// LESSR_Dec_90091234001301
// MI455X (gfx1250) — compile-verified, matrix operand order checked by reading, not run on hardware
//
#include <hip/hip_runtime.h>
#include <hip/hip_bf16.h>
#include <stdint.h>
#include <stddef.h>

namespace {

constexpr int Nn     = 32768;   // total nodes
constexpr int Dd     = 128;     // D
constexpr int Bb     = 2048;    // graphs
constexpr int Ee     = 262144;  // shortcut edges
constexpr int MaxDeg = 8;
constexpr int FeatW  = 512;     // final concat width (out2|out1|out0|emb)

typedef __attribute__((ext_vector_type(16))) __bf16 v16bf;
typedef __attribute__((ext_vector_type(8)))  float  v8f;

__device__ __forceinline__ unsigned short f2bf(float f) {
  unsigned u = __float_as_uint(f);
  unsigned r = u + 0x7FFFu + ((u >> 16) & 1u);   // round-to-nearest-even
  return (unsigned short)(r >> 16);
}
__device__ __forceinline__ float sigmoidf(float x) { return 1.0f / (1.0f + __expf(-x)); }
// order-preserving float<->uint map for atomicMax on floats
__device__ __forceinline__ unsigned fenc(float f) {
  unsigned u = __float_as_uint(f);
  return (u & 0x80000000u) ? ~u : (u | 0x80000000u);
}
__device__ __forceinline__ float fdec(unsigned e) {
  unsigned u = (e & 0x80000000u) ? (e & 0x7FFFFFFFu) : ~e;
  return __uint_as_float(u);
}

// ---------------------------------------------------------------- WMMA GEMM
// Macro-tiled: one wave computes a 32x64 block of C = 2(M) x 4(N) accumulators.
// A[M,K] bf16 row-major (lda); Bt[Nc,K] bf16 pre-transposed (ldb).
// Per K-step(32): 6 fragment loads -> 8 v_wmma_f32_16x16x32_bf16.
__global__ __launch_bounds__(256)
void gemm_bf16_wmma_mac_k(const unsigned short* __restrict__ A, int lda,
                          const unsigned short* __restrict__ Bt, int ldb,
                          float* __restrict__ C, int ldc,
                          const float* __restrict__ bias,
                          int M, int Nc, int K, int accum) {
  const int TN    = Nc >> 6;              // 64-wide column tiles
  const int tiles = (M >> 5) * TN;        // 32-tall row tiles
  const int wave  = (int)((blockIdx.x * blockDim.x + threadIdx.x) >> 5);
  if (wave >= tiles) return;
  const int tm   = wave / TN;
  const int tn   = wave % TN;
  const int lane = threadIdx.x & 31;
  const int half = lane >> 4;             // K-half selector
  const int mn   = lane & 15;             // row of A-frag / col of B-frag
  const unsigned short* Ar[2];
  const unsigned short* Br[4];
#pragma unroll
  for (int i = 0; i < 2; ++i) Ar[i] = A  + (size_t)(tm * 32 + i * 16 + mn) * lda;
#pragma unroll
  for (int j = 0; j < 4; ++j) Br[j] = Bt + (size_t)(tn * 64 + j * 16 + mn) * ldb;

  v8f acc[2][4];
#pragma unroll
  for (int i = 0; i < 2; ++i)
#pragma unroll
    for (int j = 0; j < 4; ++j)
      acc[i][j] = (v8f){0.f, 0.f, 0.f, 0.f, 0.f, 0.f, 0.f, 0.f};

  for (int kb = 0; kb < K; kb += 32) {
    union { unsigned u[8]; v16bf v; } a[2], b[4];
#pragma unroll
    for (int r = 0; r < 8; ++r) {
      // 16-bit 16x32 fragment lane layout: K = group*16 + half*8 + (r%4)*2
      const int kk = ((r >> 2) << 4) + (half << 3) + ((r & 3) << 1);
      a[0].u[r] = *(const unsigned*)(Ar[0] + kb + kk);
      a[1].u[r] = *(const unsigned*)(Ar[1] + kb + kk);
      b[0].u[r] = *(const unsigned*)(Br[0] + kb + kk);
      b[1].u[r] = *(const unsigned*)(Br[1] + kb + kk);
      b[2].u[r] = *(const unsigned*)(Br[2] + kb + kk);
      b[3].u[r] = *(const unsigned*)(Br[3] + kb + kk);
    }
    if (kb + 32 < K) {
      __builtin_prefetch(Ar[0] + kb + 32, 0, 1);   // global_prefetch_b8
      __builtin_prefetch(Ar[1] + kb + 32, 0, 1);
    }
#pragma unroll
    for (int i = 0; i < 2; ++i)
#pragma unroll
      for (int j = 0; j < 4; ++j)
        acc[i][j] = __builtin_amdgcn_wmma_f32_16x16x32_bf16(
            false, a[i].v, false, b[j].v, (short)0, acc[i][j], false, false);
  }

#pragma unroll
  for (int i = 0; i < 2; ++i)
#pragma unroll
    for (int j = 0; j < 4; ++j) {
      const int colb = tn * 64 + j * 16 + mn;
      const float bs = bias ? bias[colb] : 0.f;
#pragma unroll
      for (int r = 0; r < 8; ++r) {
        const int row = tm * 32 + i * 16 + r + 8 * half;  // C/D: m = vgpr + 8*half
        const size_t idx = (size_t)row * ldc + colb;
        float v = acc[i][j][r] + bs;
        if (accum) v += C[idx];
        C[idx] = v;
      }
    }
}

// Fallback: one wave -> one 16x16 tile (for shapes not divisible by 32/64).
__global__ __launch_bounds__(256)
void gemm_bf16_wmma_k(const unsigned short* __restrict__ A, int lda,
                      const unsigned short* __restrict__ Bt, int ldb,
                      float* __restrict__ C, int ldc,
                      const float* __restrict__ bias,
                      int M, int Nc, int K, int accum) {
  const int TN    = Nc >> 4;
  const int tiles = (M >> 4) * TN;
  const int wave  = (int)((blockIdx.x * blockDim.x + threadIdx.x) >> 5);
  if (wave >= tiles) return;
  const int tm   = wave / TN;
  const int tn   = wave % TN;
  const int lane = threadIdx.x & 31;
  const int half = lane >> 4;
  const int mn   = lane & 15;
  const unsigned short* Arow = A  + (size_t)(tm * 16 + mn) * lda;
  const unsigned short* Brow = Bt + (size_t)(tn * 16 + mn) * ldb;
  v8f acc = {0.f, 0.f, 0.f, 0.f, 0.f, 0.f, 0.f, 0.f};
  for (int kb = 0; kb < K; kb += 32) {
    union { unsigned u[8]; v16bf v; } a, b;
#pragma unroll
    for (int r = 0; r < 8; ++r) {
      const int kk = ((r >> 2) << 4) + (half << 3) + ((r & 3) << 1);
      a.u[r] = *(const unsigned*)(Arow + kb + kk);
      b.u[r] = *(const unsigned*)(Brow + kb + kk);
    }
    acc = __builtin_amdgcn_wmma_f32_16x16x32_bf16(
        false, a.v, false, b.v, (short)0, acc, false, false);
  }
#pragma unroll
  for (int r = 0; r < 8; ++r) {
    const int row = tm * 16 + r + 8 * half;
    const int col = tn * 16 + mn;
    const size_t idx = (size_t)row * ldc + col;
    float v = acc[r];
    if (bias)  v += bias[col];
    if (accum) v += C[idx];
    C[idx] = v;
  }
}

// ---------------------------------------------------------------- helpers
__global__ void wtrans_k(const float* __restrict__ W, unsigned short* __restrict__ Wt,
                         int K, int Nc) {
  int i = blockIdx.x * blockDim.x + threadIdx.x;
  if (i >= K * Nc) return;
  int k = i / Nc, n = i % Nc;
  Wt[(size_t)n * K + k] = f2bf(W[i]);   // [K,Nc] -> bf16 [Nc,K]
}

__global__ void gather_embed_k(const float* __restrict__ emb, const int* __restrict__ iid,
                               float* __restrict__ FEAT) {
  int i = blockIdx.x * blockDim.x + threadIdx.x;
  if (i >= Nn * Dd) return;
  int n = i >> 7, c = i & 127;
  FEAT[(size_t)n * FeatW + 384 + c] = emb[(size_t)iid[n] * Dd + c];
}

__global__ __launch_bounds__(256)
void colstats_k(const float* __restrict__ X, int ld, int rows,
                float* __restrict__ mu, float* __restrict__ rs) {
  const int c = blockIdx.x;
  __shared__ float s1[256], s2[256];
  float a = 0.f, b = 0.f;
  for (int r = threadIdx.x; r < rows; r += 256) {
    float v = X[(size_t)r * ld + c];
    a += v; b += v * v;
  }
  s1[threadIdx.x] = a; s2[threadIdx.x] = b;
  __syncthreads();
  for (int o = 128; o; o >>= 1) {
    if ((int)threadIdx.x < o) { s1[threadIdx.x] += s1[threadIdx.x + o];
                                s2[threadIdx.x] += s2[threadIdx.x + o]; }
    __syncthreads();
  }
  if (threadIdx.x == 0) {
    float m = s1[0] / rows;
    float var = s2[0] / rows - m * m;
    mu[c] = m; rs[c] = rsqrtf(var + 1e-5f);
  }
}

__global__ void bn_apply_k(const float* __restrict__ X, int ld,
                           const float* __restrict__ mu, const float* __restrict__ rs,
                           const float* __restrict__ g,  const float* __restrict__ b,
                           float* __restrict__ Y32, unsigned short* __restrict__ Y16,
                           int rows, int C) {
  int i = blockIdx.x * blockDim.x + threadIdx.x;
  if (i >= rows * C) return;
  int r = i / C, c = i % C;
  float v = (X[(size_t)r * ld + c] - mu[c]) * rs[c] * g[c] + b[c];
  Y32[i] = v;
  Y16[i] = f2bf(v);
}

__global__ void fillf_k(float* p, int n, float v) {
  int i = blockIdx.x * blockDim.x + threadIdx.x; if (i < n) p[i] = v;
}
__global__ void fillu16_k(unsigned short* p, int n, unsigned short v) {
  int i = blockIdx.x * blockDim.x + threadIdx.x; if (i < n) p[i] = v;
}
__global__ void fillu32_k(unsigned* p, int n, unsigned v) {
  int i = blockIdx.x * blockDim.x + threadIdx.x; if (i < n) p[i] = v;
}
__global__ void fill_view_k(float* p, int ld, int rows, int C, float v) {
  int i = blockIdx.x * blockDim.x + threadIdx.x;
  if (i >= rows * C) return;
  int r = i / C, c = i % C;
  p[(size_t)r * ld + c] = v;
}
__global__ void cvt16_k(const float* __restrict__ X, unsigned short* __restrict__ Y, int n) {
  int i = blockIdx.x * blockDim.x + threadIdx.x; if (i < n) Y[i] = f2bf(X[i]);
}

__global__ void gather_mb_k(const unsigned short* __restrict__ XBN16,
                            const int* __restrict__ nbr, int t,
                            unsigned short* __restrict__ MT, int rows, int din) {
  int i = blockIdx.x * blockDim.x + threadIdx.x;
  if (i >= rows * din) return;
  int n = i / din, c = i % din;
  MT[i] = XBN16[(size_t)nbr[n * MaxDeg + t] * din + c];
}

__global__ void gru_step_k(const float* __restrict__ XW, const float* __restrict__ HW,
                           float* __restrict__ H, unsigned short* __restrict__ H16,
                           const int* __restrict__ nbr_len, int t, int rows, int din) {
  int i = blockIdx.x * blockDim.x + threadIdx.x;
  if (i >= rows * din) return;
  int n = i / din, c = i % din;
  const int d3 = 3 * din;
  const size_t base = (size_t)n * d3;
  float xr = XW[base + c], xz = XW[base + din + c], xn = XW[base + 2 * din + c];
  float hr = HW[base + c], hz = HW[base + din + c], hn = HW[base + 2 * din + c];
  float r  = sigmoidf(xr + hr);
  float z  = sigmoidf(xz + hz);
  float nn = tanhf(xn + r * hn);
  float ho = H[i];
  float h  = (t < nbr_len[n]) ? (1.f - z) * nn + z * ho : ho;
  H[i]   = h;
  H16[i] = f2bf(h);
}

__global__ void prelu_k(float* __restrict__ X, int ld, int rows, int C,
                        const float* __restrict__ a) {
  int i = blockIdx.x * blockDim.x + threadIdx.x;
  if (i >= rows * C) return;
  int r = i / C, c = i % C;
  float v = X[(size_t)r * ld + c];
  if (v < 0.f) X[(size_t)r * ld + c] = a[c] * v;
}

// sigmoid(q[src]+k[dst]) . We  -> one wave per edge
__global__ __launch_bounds__(256)
void edge_score_k(const float* __restrict__ Q, const float* __restrict__ Kk,
                  const int* __restrict__ src, const int* __restrict__ dst,
                  const float* __restrict__ We, float* __restrict__ Ev, int E) {
  int w = (int)((blockIdx.x * blockDim.x + threadIdx.x) >> 5);
  int lane = threadIdx.x & 31;
  if (w >= E) return;
  int s = src[w], d = dst[w];
  float acc = 0.f;
  for (int c = lane; c < Dd; c += 32)
    acc += We[c] * sigmoidf(Q[(size_t)s * Dd + c] + Kk[(size_t)d * Dd + c]);
#pragma unroll
  for (int o = 16; o; o >>= 1) acc += __shfl_xor(acc, o);
  if (lane == 0) Ev[w] = acc;
}

__global__ void edge_max_k(const float* __restrict__ Ev, const int* __restrict__ dst,
                           unsigned* __restrict__ key, int E) {
  int i = blockIdx.x * blockDim.x + threadIdx.x;
  if (i < E) atomicMax(&key[dst[i]], fenc(Ev[i]));
}
__global__ void edge_expsum_k(float* __restrict__ Ev, const int* __restrict__ dst,
                              const unsigned* __restrict__ key, float* __restrict__ sum,
                              int E) {
  int i = blockIdx.x * blockDim.x + threadIdx.x;
  if (i >= E) return;
  float m  = fdec(key[dst[i]]);
  float ex = __expf(Ev[i] - m);
  Ev[i] = ex;
  atomicAdd(&sum[dst[i]], ex);
}
__global__ __launch_bounds__(256)
void edge_agg_k(const float* __restrict__ Ev, const float* __restrict__ sum,
                const int* __restrict__ src, const int* __restrict__ dst,
                const float* __restrict__ V, float* __restrict__ Cout, int ldc, int E) {
  int w = (int)((blockIdx.x * blockDim.x + threadIdx.x) >> 5);
  int lane = threadIdx.x & 31;
  if (w >= E) return;
  int s = src[w], d = dst[w];
  float alpha = Ev[w] / sum[d];
  for (int c = lane; c < Dd; c += 32)
    atomicAdd(&Cout[(size_t)d * ldc + c], V[(size_t)s * Dd + c] * alpha);
}

// readout node scores: sigmoid(fu[n]+fv[graph(n)]) . We
__global__ __launch_bounds__(256)
void node_score_k(const float* __restrict__ FU, const float* __restrict__ FVL,
                  const int* __restrict__ gid, const float* __restrict__ We,
                  float* __restrict__ EN, int rows) {
  int w = (int)((blockIdx.x * blockDim.x + threadIdx.x) >> 5);
  int lane = threadIdx.x & 31;
  if (w >= rows) return;
  int g = gid[w];
  float acc = 0.f;
  for (int c = lane; c < Dd; c += 32)
    acc += We[c] * sigmoidf(FU[(size_t)w * Dd + c] + FVL[(size_t)g * Dd + c]);
#pragma unroll
  for (int o = 16; o; o >>= 1) acc += __shfl_xor(acc, o);
  if (lane == 0) EN[w] = acc;
}

// per-graph (16 contiguous nodes) softmax + weighted feature sum -> [B,512]
__global__ __launch_bounds__(256)
void graph_agg_k(const float* __restrict__ EN, const float* __restrict__ XBN,
                 float* __restrict__ OUT) {
  int g = blockIdx.x;
  __shared__ float al[16];
  if (threadIdx.x == 0) {
    float m = -1e30f;
    for (int j = 0; j < 16; ++j) m = fmaxf(m, EN[g * 16 + j]);
    float tmp[16]; float s = 0.f;
    for (int j = 0; j < 16; ++j) { tmp[j] = __expf(EN[g * 16 + j] - m); s += tmp[j]; }
    for (int j = 0; j < 16; ++j) al[j] = tmp[j] / s;
  }
  __syncthreads();
  for (int d = threadIdx.x; d < FeatW; d += 256) {
    float acc = 0.f;
    for (int j = 0; j < 16; ++j)
      acc += XBN[(size_t)(g * 16 + j) * FeatW + d] * al[j];
    OUT[(size_t)g * FeatW + d] = acc;
  }
}

__global__ void gather_rows16_k(const unsigned short* __restrict__ X, int ld,
                                const int* __restrict__ idx,
                                unsigned short* __restrict__ Y, int rows, int C) {
  int i = blockIdx.x * blockDim.x + threadIdx.x;
  if (i >= rows * C) return;
  int r = i / C, c = i % C;
  Y[i] = X[(size_t)idx[r] * ld + c];
}

__global__ void build_sr16_k(const float* __restrict__ FEAT, const int* __restrict__ last,
                             const float* __restrict__ SRG, unsigned short* __restrict__ SR16) {
  int i = blockIdx.x * blockDim.x + threadIdx.x;
  if (i >= Bb * 640) return;
  int b = i / 640, c = i % 640;
  float v = (c < FeatW) ? FEAT[(size_t)last[b] * FeatW + c]
                        : SRG[(size_t)b * Dd + (c - FeatW)];
  SR16[i] = f2bf(v);
}

} // anonymous namespace

extern "C" void kernel_launch(void* const* d_in, const int* in_sizes, int n_in,
                              void* d_out, int out_size, void* d_ws, size_t ws_size,
                              hipStream_t stream) {
  (void)in_sizes; (void)n_in; (void)out_size; (void)ws_size;

  // ---- inputs (setup_inputs dict order, params flattened in order)
  const int*   iid       = (const int*)d_in[0];
  const int*   nbr_idx   = (const int*)d_in[1];
  const int*   nbr_len   = (const int*)d_in[2];
  const int*   src       = (const int*)d_in[3];
  const int*   dst       = (const int*)d_in[4];
  const int*   graph_id  = (const int*)d_in[5];
  const int*   last_nodes= (const int*)d_in[6];
  const float* emb       = (const float*)d_in[7];
  const float* l0_bn_g   = (const float*)d_in[8];
  const float* l0_bn_b   = (const float*)d_in[9];
  const float* l0_Wi     = (const float*)d_in[10];
  const float* l0_Wh     = (const float*)d_in[11];
  const float* l0_bi     = (const float*)d_in[12];
  const float* l0_bh     = (const float*)d_in[13];
  const float* l0_Wself  = (const float*)d_in[14];
  const float* l0_Wneigh = (const float*)d_in[15];
  const float* l0_prelu  = (const float*)d_in[16];
  const float* l1_bn_g   = (const float*)d_in[17];
  const float* l1_bn_b   = (const float*)d_in[18];
  const float* l1_Wq     = (const float*)d_in[19];
  const float* l1_bq     = (const float*)d_in[20];
  const float* l1_Wk     = (const float*)d_in[21];
  const float* l1_Wv     = (const float*)d_in[22];
  const float* l1_We     = (const float*)d_in[23];
  const float* l1_prelu  = (const float*)d_in[24];
  const float* l2_bn_g   = (const float*)d_in[25];
  const float* l2_bn_b   = (const float*)d_in[26];
  const float* l2_Wi     = (const float*)d_in[27];
  const float* l2_Wh     = (const float*)d_in[28];
  const float* l2_bi     = (const float*)d_in[29];
  const float* l2_bh     = (const float*)d_in[30];
  const float* l2_Wself  = (const float*)d_in[31];
  const float* l2_Wneigh = (const float*)d_in[32];
  const float* l2_prelu  = (const float*)d_in[33];
  const float* ro_bn_g   = (const float*)d_in[34];
  const float* ro_bn_b   = (const float*)d_in[35];
  const float* ro_Wu     = (const float*)d_in[36];
  const float* ro_Wv     = (const float*)d_in[37];
  const float* ro_bv     = (const float*)d_in[38];
  const float* ro_We     = (const float*)d_in[39];
  const float* ro_Wout   = (const float*)d_in[40];
  const float* ro_prelu  = (const float*)d_in[41];
  const float* W_sr      = (const float*)d_in[42];
  const float* W_rf      = (const float*)d_in[43];
  const float* b_rf      = (const float*)d_in[44];

  float* out = (float*)d_out;   // dec [2048,2048] then logits [2048,128]

  // ---- workspace bump allocator
  char* wsb = (char*)d_ws;
  size_t off = 0;
  auto alloc = [&](size_t bytes) -> void* {
    off = (off + 255) & ~(size_t)255;
    void* p = wsb + off;
    off += bytes;
    return p;
  };
  float*          FEAT   = (float*)alloc((size_t)Nn * FeatW * 4);
  float*          XBN32  = (float*)alloc((size_t)Nn * FeatW * 4);
  unsigned short* XBN16  = (unsigned short*)alloc((size_t)Nn * FeatW * 2);
  unsigned short* MT16   = (unsigned short*)alloc((size_t)Nn * 384 * 2);
  float*          Hbuf   = (float*)alloc((size_t)Nn * 384 * 4);
  unsigned short* H16    = (unsigned short*)alloc((size_t)Nn * 384 * 2);
  float*          XW     = (float*)alloc((size_t)Nn * 1152 * 4);
  float*          HW     = (float*)alloc((size_t)Nn * 1152 * 4);
  float*          Qb     = (float*)alloc((size_t)Nn * Dd * 4);
  float*          Kb     = (float*)alloc((size_t)Nn * Dd * 4);
  float*          Vb     = (float*)alloc((size_t)Nn * Dd * 4);
  float*          Ev     = (float*)alloc((size_t)Ee * 4);
  unsigned*       NodeKey= (unsigned*)alloc((size_t)Nn * 4);
  float*          NodeSum= (float*)alloc((size_t)Nn * 4);
  float*          FU     = (float*)alloc((size_t)Nn * Dd * 4);
  float*          FVL    = (float*)alloc((size_t)Bb * Dd * 4);
  float*          EN     = (float*)alloc((size_t)Nn * 4);
  float*          SRGp   = (float*)alloc((size_t)Bb * FeatW * 4);
  unsigned short* SRGp16 = (unsigned short*)alloc((size_t)Bb * FeatW * 2);
  float*          SRG    = (float*)alloc((size_t)Bb * Dd * 4);
  unsigned short* XL16   = (unsigned short*)alloc((size_t)Bb * FeatW * 2);
  unsigned short* SR16   = (unsigned short*)alloc((size_t)Bb * 640 * 2);
  float*          MU     = (float*)alloc(FeatW * 4);
  float*          RS     = (float*)alloc(FeatW * 4);

  auto cdiv = [](int a, int b) { return (a + b - 1) / b; };

  auto wt = [&](const float* W, int K, int Nc) -> unsigned short* {
    unsigned short* p = (unsigned short*)alloc((size_t)K * Nc * 2);
    int n = K * Nc;
    wtrans_k<<<cdiv(n, 256), 256, 0, stream>>>(W, p, K, Nc);
    return p;
  };
  auto gemm = [&](const unsigned short* A, int lda, const unsigned short* Bt, int ldb,
                  float* C, int ldc, const float* bias, int M, int Nc, int K, int accum) {
    if ((M % 32) == 0 && (Nc % 64) == 0) {
      int tiles = (M / 32) * (Nc / 64);
      gemm_bf16_wmma_mac_k<<<cdiv(tiles, 8), 256, 0, stream>>>(A, lda, Bt, ldb, C, ldc,
                                                               bias, M, Nc, K, accum);
    } else {
      int tiles = (M / 16) * (Nc / 16);
      gemm_bf16_wmma_k<<<cdiv(tiles, 8), 256, 0, stream>>>(A, lda, Bt, ldb, C, ldc,
                                                           bias, M, Nc, K, accum);
    }
  };
  auto bn = [&](int coloff, int din) {
    colstats_k<<<din, 256, 0, stream>>>(FEAT + coloff, FeatW, Nn, MU, RS);
    int n = Nn * din;
    bn_apply_k<<<cdiv(n, 256), 256, 0, stream>>>(
        FEAT + coloff, FeatW, MU, RS,
        (coloff == 384) ? l0_bn_g : (coloff == 256) ? l1_bn_g : (coloff == 128) ? l2_bn_g : ro_bn_g,
        (coloff == 384) ? l0_bn_b : (coloff == 256) ? l1_bn_b : (coloff == 128) ? l2_bn_b : ro_bn_b,
        XBN32, XBN16, Nn, din);
  };

  // ---- stage 0: embedding gather -> FEAT[:,384:512]
  gather_embed_k<<<cdiv(Nn * Dd, 256), 256, 0, stream>>>(emb, iid, FEAT);

  // ---- EOPA layer runner
  auto run_eopa = [&](int din, int coloff_in, int coloff_out,
                      const float* Wi, const float* Wh, const float* bi, const float* bh,
                      const float* Wself, const float* Wneigh, const float* prelu_a) {
    const int d3 = 3 * din;
    unsigned short* Wi_t = wt(Wi, din, d3);
    unsigned short* Wh_t = wt(Wh, din, d3);
    unsigned short* Ws_t = wt(Wself, din, Dd);
    unsigned short* Wn_t = wt(Wneigh, din, Dd);
    bn(coloff_in, din);
    fillf_k<<<cdiv(Nn * din, 256), 256, 0, stream>>>(Hbuf, Nn * din, 0.f);
    fillu16_k<<<cdiv(Nn * din, 256), 256, 0, stream>>>(H16, Nn * din, 0);
    for (int t = 0; t < MaxDeg; ++t) {
      gather_mb_k<<<cdiv(Nn * din, 256), 256, 0, stream>>>(XBN16, nbr_idx, t, MT16, Nn, din);
      gemm(MT16, din, Wi_t, din, XW, d3, bi, Nn, d3, din, 0);
      gemm(H16,  din, Wh_t, din, HW, d3, bh, Nn, d3, din, 0);
      gru_step_k<<<cdiv(Nn * din, 256), 256, 0, stream>>>(XW, HW, Hbuf, H16, nbr_len, t, Nn, din);
    }
    gemm(XBN16, din, Ws_t, din, FEAT + coloff_out, FeatW, nullptr, Nn, Dd, din, 0);
    gemm(H16,  din, Wn_t, din, FEAT + coloff_out, FeatW, nullptr, Nn, Dd, din, 1);
    prelu_k<<<cdiv(Nn * Dd, 256), 256, 0, stream>>>(FEAT + coloff_out, FeatW, Nn, Dd, prelu_a);
  };

  // ---- layer 0: EOPA (din=128), input cols 384.., output cols 256..
  run_eopa(128, 384, 256, l0_Wi, l0_Wh, l0_bi, l0_bh, l0_Wself, l0_Wneigh, l0_prelu);

  // ---- layer 1: SGAT (din=256), input cols 256.., output cols 128..
  {
    unsigned short* Wq_t = wt(l1_Wq, 256, Dd);
    unsigned short* Wk_t = wt(l1_Wk, 256, Dd);
    unsigned short* Wv_t = wt(l1_Wv, 256, Dd);
    bn(256, 256);
    gemm(XBN16, 256, Wq_t, 256, Qb, Dd, l1_bq,  Nn, Dd, 256, 0);
    gemm(XBN16, 256, Wk_t, 256, Kb, Dd, nullptr, Nn, Dd, 256, 0);
    gemm(XBN16, 256, Wv_t, 256, Vb, Dd, nullptr, Nn, Dd, 256, 0);
    edge_score_k<<<cdiv(Ee * 32, 256), 256, 0, stream>>>(Qb, Kb, src, dst, l1_We, Ev, Ee);
    fillu32_k<<<cdiv(Nn, 256), 256, 0, stream>>>(NodeKey, Nn, 0u);
    edge_max_k<<<cdiv(Ee, 256), 256, 0, stream>>>(Ev, dst, NodeKey, Ee);
    fillf_k<<<cdiv(Nn, 256), 256, 0, stream>>>(NodeSum, Nn, 0.f);
    edge_expsum_k<<<cdiv(Ee, 256), 256, 0, stream>>>(Ev, dst, NodeKey, NodeSum, Ee);
    fill_view_k<<<cdiv(Nn * Dd, 256), 256, 0, stream>>>(FEAT + 128, FeatW, Nn, Dd, 0.f);
    edge_agg_k<<<cdiv(Ee * 32, 256), 256, 0, stream>>>(Ev, NodeSum, src, dst, Vb,
                                                       FEAT + 128, FeatW, Ee);
    prelu_k<<<cdiv(Nn * Dd, 256), 256, 0, stream>>>(FEAT + 128, FeatW, Nn, Dd, l1_prelu);
  }

  // ---- layer 2: EOPA (din=384), input cols 128.., output cols 0..
  run_eopa(384, 128, 0, l2_Wi, l2_Wh, l2_bi, l2_bh, l2_Wself, l2_Wneigh, l2_prelu);

  // ---- readout (din=512)
  {
    unsigned short* Wu_t = wt(ro_Wu, FeatW, Dd);
    unsigned short* Wv_t = wt(ro_Wv, FeatW, Dd);
    unsigned short* Wo_t = wt(ro_Wout, FeatW, Dd);
    bn(0, FeatW);
    gemm(XBN16, FeatW, Wu_t, FeatW, FU, Dd, nullptr, Nn, Dd, FeatW, 0);
    gather_rows16_k<<<cdiv(Bb * FeatW, 256), 256, 0, stream>>>(XBN16, FeatW, last_nodes,
                                                               XL16, Bb, FeatW);
    gemm(XL16, FeatW, Wv_t, FeatW, FVL, Dd, ro_bv, Bb, Dd, FeatW, 0);
    node_score_k<<<cdiv(Nn * 32, 256), 256, 0, stream>>>(FU, FVL, graph_id, ro_We, EN, Nn);
    graph_agg_k<<<Bb, 256, 0, stream>>>(EN, XBN32, SRGp);
    cvt16_k<<<cdiv(Bb * FeatW, 256), 256, 0, stream>>>(SRGp, SRGp16, Bb * FeatW);
    gemm(SRGp16, FeatW, Wo_t, FeatW, SRG, Dd, nullptr, Bb, Dd, FeatW, 0);
    prelu_k<<<cdiv(Bb * Dd, 256), 256, 0, stream>>>(SRG, Dd, Bb, Dd, ro_prelu);
  }

  // ---- sr = [feat[last_nodes] | sr_g], final GEMMs -> d_out
  build_sr16_k<<<cdiv(Bb * 640, 256), 256, 0, stream>>>(FEAT, last_nodes, SRG, SR16);
  {
    unsigned short* Wrf_t = wt(W_rf, 640, 2048);
    unsigned short* Wsr_t = wt(W_sr, 640, Dd);
    gemm(SR16, 640, Wrf_t, 640, out, 2048, b_rf, Bb, 2048, 640, 0);               // dec
    gemm(SR16, 640, Wsr_t, 640, out + (size_t)Bb * 2048, Dd, nullptr, Bb, Dd, 640, 0); // logits
  }
}
